// GEN_66924180407021
// MI455X (gfx1250) — compile-verified, matrix operand order checked by reading, not run on hardware
//
#include <hip/hip_runtime.h>

// ---------------------------------------------------------------------------
// Edge MLP link-prediction loss on gfx1250 (MI455X).
// Core GEMM: [E,256] x [256,512] in bf16 via v_wmma_f32_16x16x32_bf16.
// ---------------------------------------------------------------------------

typedef __attribute__((ext_vector_type(16))) __bf16 v16bf;
typedef __attribute__((ext_vector_type(8)))  float  v8f;
typedef __attribute__((ext_vector_type(4)))  unsigned int v4u;

#define DIM 256
#define HID 512
#define M_TILE 64
#define LDS_STRIDE (DIM + 8)   // in bf16 units; +16B pad -> conflict-free ds_load_b128

__device__ __forceinline__ unsigned short f2bf(float x) {
    union { float f; unsigned int u; } c; c.f = x;
    unsigned int u = c.u;
    return (unsigned short)((u + 0x7FFFu + ((u >> 16) & 1u)) >> 16);  // RNE
}

__global__ void GEN_zero_out(float* out) { out[0] = 0.0f; }

// W1 [K=256][N=512] f32 (row-major) -> W1^T [N=512][K=256] bf16, so B-matrix
// fragments load with the same per-lane layout as A fragments.
__global__ void GEN_w1_transpose_bf16(const float* __restrict__ W1,
                                      unsigned short* __restrict__ w1t) {
    int id = blockIdx.x * blockDim.x + threadIdx.x;   // 0 .. 256*512-1
    int k = id & (DIM - 1);
    int n = id >> 8;
    w1t[n * DIM + k] = f2bf(W1[k * HID + n]);
}

__global__ void __launch_bounds__(256)
GEN_edge_mlp_loss(const float* __restrict__ x_s,
                  const int*   __restrict__ edges,
                  const unsigned short* __restrict__ w1t,
                  const float* __restrict__ b1,
                  const float* __restrict__ W2,
                  const float* __restrict__ b2,
                  float* __restrict__ out,
                  int n_edges)
{
    __shared__ unsigned short sA[M_TILE * LDS_STRIDE]; // 64 x 264 bf16 = 33 KB
    __shared__ float sLogit[M_TILE];
    __shared__ float sSum;

    const int t  = threadIdx.x;
    const int e0 = blockIdx.x * M_TILE;

    if (t < M_TILE) sLogit[t] = 0.0f;
    if (t == 0) sSum = 0.0f;

    // ---- Gather phase: src*dst product -> bf16 tile in LDS ----------------
    const int* esrc = edges;
    const int* edst = edges + n_edges;
    #pragma unroll
    for (int i = 0; i < (M_TILE * (DIM / 4)) / 256; ++i) {   // 16 iters
        int flat = i * 256 + t;
        int e  = flat >> 6;        // edge within tile (64 float4 per row)
        int c4 = flat & 63;        // float4 column
        int ge = e0 + e;
        float4 s = make_float4(0.f, 0.f, 0.f, 0.f);
        float4 d = make_float4(0.f, 0.f, 0.f, 0.f);
        if (ge < n_edges) {
            int si = esrc[ge];
            int di = edst[ge];
            s = *(const float4*)(x_s + (size_t)si * DIM + c4 * 4);
            d = *(const float4*)(x_s + (size_t)di * DIM + c4 * 4);
        }
        unsigned int lo = (unsigned int)f2bf(s.x * d.x) |
                          ((unsigned int)f2bf(s.y * d.y) << 16);
        unsigned int hi = (unsigned int)f2bf(s.z * d.z) |
                          ((unsigned int)f2bf(s.w * d.w) << 16);
        *(uint2*)&sA[e * LDS_STRIDE + c4 * 4] = make_uint2(lo, hi);
    }
    __syncthreads();

    // ---- GEMM phase: 64 edges x 64 hidden per wave, K = 256 ---------------
    const int wave  = t >> 5;
    const int lane  = t & 31;
    const int r     = lane & 15;   // row within 16x16 tile (M for A, N for B)
    const int hh    = lane >> 4;   // lane half -> K sub-offset per ISA layout
    const int nbase = wave * 64;   // this wave's hidden-dim slice

    union FragBF { v16bf v; v4u q[2]; };

    v8f acc[4][4] = {};            // 4 M-tiles x 4 N-tiles, f32 accum

    const unsigned short* bw = w1t + (size_t)nbase * DIM;

    for (int ks = 0; ks < DIM / 32; ++ks) {
        FragBF a[4];
        #pragma unroll
        for (int mt = 0; mt < 4; ++mt) {
            // 16-bit A 16x32 layout: each lane-half holds 8 contiguous K's
            const unsigned short* p =
                &sA[(mt * 16 + r) * LDS_STRIDE + ks * 32 + hh * 8];
            a[mt].q[0] = *(const v4u*)(p);        // K base +  0 / + 8
            a[mt].q[1] = *(const v4u*)(p + 16);   // K base + 16 / +24
        }
        #pragma unroll
        for (int nt = 0; nt < 4; ++nt) {
            FragBF bfrag;
            const unsigned short* q =
                bw + (size_t)(nt * 16 + r) * DIM + ks * 32 + hh * 8;
            bfrag.q[0] = *(const v4u*)(q);
            bfrag.q[1] = *(const v4u*)(q + 16);
            #pragma unroll
            for (int mt = 0; mt < 4; ++mt) {
                acc[mt][nt] = __builtin_amdgcn_wmma_f32_16x16x32_bf16(
                    false, a[mt].v, false, bfrag.v,
                    (short)0, acc[mt][nt], false, false);
            }
        }
    }

    // ---- Epilogue: bias + ReLU + dot(W2), reduce to per-edge logit --------
    float b1v[4], w2v[4];
    #pragma unroll
    for (int nt = 0; nt < 4; ++nt) {
        int n = nbase + nt * 16 + r;
        b1v[nt] = b1[n];
        w2v[nt] = W2[n];
    }
    #pragma unroll
    for (int mt = 0; mt < 4; ++mt) {
        #pragma unroll
        for (int j = 0; j < 8; ++j) {
            // C layout: lane (mod 16) = N, VGPR j = M (j / j+8 per half)
            float v = 0.0f;
            #pragma unroll
            for (int nt = 0; nt < 4; ++nt) {
                float hpre = acc[mt][nt][j] + b1v[nt];
                v += (hpre > 0.0f ? hpre : 0.0f) * w2v[nt];
            }
            // reduce across the 16 N-lanes of each half-wave
            v += __shfl_xor(v, 1);
            v += __shfl_xor(v, 2);
            v += __shfl_xor(v, 4);
            v += __shfl_xor(v, 8);
            if (r == 0) {
                int edge = mt * 16 + j + (hh << 3);
                atomicAdd(&sLogit[edge], v);
            }
        }
    }
    __syncthreads();

    // ---- Loss: sigmoid + -log(p+eps), block reduce, global accumulate ----
    if (t < M_TILE) {
        float c = 0.0f;
        int ge = e0 + t;
        if (ge < n_edges) {
            float z = sLogit[t] + b2[0];
            float p = 1.0f / (1.0f + __expf(-z));
            c = -__logf(p + 1e-15f);
        }
        atomicAdd(&sSum, c);
    }
    __syncthreads();
    if (t == 0) atomicAdd(out, sSum * (1.0f / (float)n_edges));
}

extern "C" void kernel_launch(void* const* d_in, const int* in_sizes, int n_in,
                              void* d_out, int out_size, void* d_ws, size_t ws_size,
                              hipStream_t stream) {
    // Input order: x_f(0), mask_nodes(1), x_s(2), edges(3), W1(4), b1(5), W2(6), b2(7)
    // x_f and mask_nodes are unused by the reference loss.
    const float* x_s  = (const float*)d_in[2];
    const int*   edges = (const int*)d_in[3];
    const float* W1   = (const float*)d_in[4];
    const float* b1   = (const float*)d_in[5];
    const float* W2   = (const float*)d_in[6];
    const float* b2   = (const float*)d_in[7];
    float* out = (float*)d_out;

    int n_edges = in_sizes[3] / 2;   // edges is [2, E]

    unsigned short* w1t = (unsigned short*)d_ws;  // 512*256*2 = 256 KB

    GEN_zero_out<<<1, 1, 0, stream>>>(out);
    GEN_w1_transpose_bf16<<<(DIM * HID) / 256, 256, 0, stream>>>(W1, w1t);

    int nblocks = (n_edges + M_TILE - 1) / M_TILE;
    GEN_edge_mlp_loss<<<nblocks, 256, 0, stream>>>(x_s, edges, w1t, b1, W2, b2,
                                                   out, n_edges);
}
